// Attention_51196010168667
// MI455X (gfx1250) — matrix operand order repaired and checked by reading, not compiled, not run on hardware
//
#include <hip/hip_runtime.h>
#include <stdint.h>

typedef __bf16 bf16;
typedef __attribute__((ext_vector_type(16))) __bf16 bf16x16;
typedef __attribute__((ext_vector_type(8)))  float  f32x8;
typedef __attribute__((ext_vector_type(4)))  int    v4i;

#define HEADS 8
#define HD 64
#define NSP 1024   // H*W
#define CCH 512
#define BB 8

#define GLOBAL_AS __attribute__((address_space(1)))
#define LDS_AS    __attribute__((address_space(3)))

#if defined(__has_builtin)
#if __has_builtin(__builtin_amdgcn_global_load_async_to_lds_b128) && \
    __has_builtin(__builtin_amdgcn_s_wait_asynccnt)
#define USE_ASYNC 1
#endif
#endif
#ifndef USE_ASYNC
#define USE_ASYNC 0
#endif

#if USE_ASYNC
__device__ __forceinline__ void async_cp16(const void* g, void* l) {
  __builtin_amdgcn_global_load_async_to_lds_b128(
      (GLOBAL_AS v4i*)g, (LDS_AS v4i*)l, 0, 0);
}
#endif

__device__ __forceinline__ bf16 tobf(float f) { return (bf16)f; }

__device__ __forceinline__ uint32_t packbf(float lo, float hi) {
  union { bf16 h[2]; uint32_t u; } r;
  r.h[0] = (bf16)lo; r.h[1] = (bf16)hi;
  return r.u;
}

// ---------------------------------------------------------------- GroupNorm
__global__ void zero_stats(float* s) {
  if (threadIdx.x < 128) s[threadIdx.x] = 0.f;
}

// grid 512 = b(8) * g(8) * slice(8); block 256
__global__ void gn_reduce(const float* __restrict__ x, float* __restrict__ stats) {
  int blk = blockIdx.x;
  int slice = blk & 7, g = (blk >> 3) & 7, b = blk >> 6;
  const float* base = x + ((size_t)(b * CCH + g * 64)) * NSP;
  int t = threadIdx.x;
  float s1 = 0.f, s2 = 0.f;
#pragma unroll 4
  for (int i = 0; i < 32; ++i) {
    float v = base[slice * 8192 + i * 256 + t];
    s1 += v; s2 += v * v;
  }
#pragma unroll
  for (int m = 16; m >= 1; m >>= 1) {
    s1 += __shfl_xor(s1, m, 32);
    s2 += __shfl_xor(s2, m, 32);
  }
  __shared__ float sh1[8], sh2[8];
  int wave = t >> 5;
  if ((t & 31) == 0) { sh1[wave] = s1; sh2[wave] = s2; }
  __syncthreads();
  if (t == 0) {
    float a1 = 0.f, a2 = 0.f;
    for (int w = 0; w < 8; ++w) { a1 += sh1[w]; a2 += sh2[w]; }
    atomicAdd(&stats[(b * 8 + g) * 2 + 0], a1);
    atomicAdd(&stats[(b * 8 + g) * 2 + 1], a2);
  }
}

// grid 1024 = b(8) * ct(8) * nt(16); block 256. Writes xn[b][n][c] bf16.
__global__ void gn_norm(const float* __restrict__ x, const float* __restrict__ gamma,
                        const float* __restrict__ beta, const float* __restrict__ stats,
                        bf16* __restrict__ xn) {
  int blk = blockIdx.x;
  int nt = blk & 15, ct = (blk >> 4) & 7, b = blk >> 7;
  int g = ct;  // 64-channel tile == one group
  float s1 = stats[(b * 8 + g) * 2 + 0];
  float s2 = stats[(b * 8 + g) * 2 + 1];
  float mean = s1 * (1.f / 65536.f);
  float var  = s2 * (1.f / 65536.f) - mean * mean;
  float rs = rsqrtf(var + 1e-5f);
  __shared__ __align__(16) bf16 tile[64][72];
  int t = threadIdx.x;
  const float* xb = x + ((size_t)(b * CCH + ct * 64)) * NSP + nt * 64;
#pragma unroll
  for (int i = 0; i < 16; ++i) {
    int flat = i * 256 + t;
    int cc = flat >> 6, nn = flat & 63;
    float v = xb[(size_t)cc * NSP + nn];
    float y = (v - mean) * rs * gamma[ct * 64 + cc] + beta[ct * 64 + cc];
    tile[nn][cc] = tobf(y);
  }
  __syncthreads();
  bf16* xo = xn + ((size_t)b * NSP + nt * 64) * CCH + ct * 64;
#pragma unroll
  for (int i = 0; i < 16; ++i) {
    int flat = i * 256 + t;
    int nn = flat >> 6, cc = flat & 63;
    xo[(size_t)nn * CCH + cc] = tile[nn][cc];
  }
}

// ---------------------------------------------------------------- QKV GEMM
// grid 3072 = b(8) * mt(16) * ot(24); block 128 (4 waves)
__global__ void qkv_gemm(const bf16* __restrict__ xn, const float* __restrict__ wq,
                         bf16* __restrict__ q, bf16* __restrict__ k,
                         bf16* __restrict__ v) {
  int blk = blockIdx.x;
  int ot = blk % 24, mt = (blk / 24) & 15, b = blk / (24 * 16);
  int o0 = ot * 64, m0 = mt * 64;
  __shared__ __align__(16) bf16 XS[64][40];
  __shared__ __align__(16) bf16 WS[64][40];
  int t = threadIdx.x, lane = t & 31, wave = t >> 5;
  int l15 = lane & 15, half = lane >> 4;
  f32x8 acc[4] = {};
  const bf16* xrow = xn + ((size_t)b * NSP + m0) * CCH;
  for (int kt = 0; kt < 16; ++kt) {
    __syncthreads();
#if USE_ASYNC
    {  // async stage A tile (64 x 32 bf16) : 256 x 16B chunks
#pragma unroll
      for (int i = 0; i < 2; ++i) {
        int c = i * 128 + t;
        int r = c >> 2, qd = c & 3;
        async_cp16(xrow + (size_t)r * CCH + kt * 32 + qd * 8, &XS[r][qd * 8]);
      }
    }
#else
    {  // stage A tile (64 x 32 bf16) as dwords
      const uint32_t* src = (const uint32_t*)xrow + kt * 16;
#pragma unroll
      for (int i = 0; i < 8; ++i) {
        int flat = i * 128 + t;
        int r = flat >> 4, cd = flat & 15;
        ((uint32_t*)&XS[r][0])[cd] = src[(size_t)r * 256 + cd];
      }
    }
#endif
    {  // stage W tile (64 o-rows x 32 c), f32 -> packed bf16
      const float* wsrc = wq + (size_t)o0 * CCH + kt * 32;
#pragma unroll
      for (int i = 0; i < 8; ++i) {
        int flat = i * 128 + t;  // 1024 dword pairs
        int r = flat >> 4, cp = flat & 15;
        float2 pv = *(const float2*)(wsrc + (size_t)r * CCH + cp * 2);
        ((uint32_t*)&WS[r][0])[cp] = packbf(pv.x, pv.y);
      }
    }
#if USE_ASYNC
    __builtin_amdgcn_s_wait_asynccnt(0);
#endif
    __syncthreads();
    int ah = half * 16;
    int am = wave * 16 + l15;
    union { bf16x16 v; uint4 u[2]; } A;
    A.u[0] = *(const uint4*)&XS[am][ah];
    A.u[1] = *(const uint4*)&XS[am][ah + 8];
    int bh = half * 16;
#pragma unroll
    for (int grp = 0; grp < 4; ++grp) {
      union { bf16x16 v; uint4 u[2]; } Bm;
      const uint4* p = (const uint4*)&WS[grp * 16 + l15][bh];
      Bm.u[0] = p[0]; Bm.u[1] = p[1];
      acc[grp] = __builtin_amdgcn_wmma_f32_16x16x32_bf16(
          false, A.v, false, Bm.v, (short)0, acc[grp], false, false);
    }
  }
  // epilogue: o-tile (64-aligned) is a single (sel, head) slice
  int sel = o0 >> 9;
  int head = (o0 >> 6) & 7;
  bf16* dst = (sel == 0 ? q : (sel == 1 ? k : v)) +
              ((size_t)b * HEADS + head) * NSP * HD;
#pragma unroll
  for (int grp = 0; grp < 4; ++grp) {
    int d = grp * 16 + l15;
#pragma unroll
    for (int r = 0; r < 8; ++r) {
      int n = m0 + wave * 16 + r + half * 8;
      dst[(size_t)n * HD + d] = tobf(acc[grp][r]);
    }
  }
}

// ---------------------------------------------------------------- attention
// grid 1024 = b(8) * h(8) * rowtile(16); block 128 (4 waves x 16 Q rows)
__global__ void attn_kernel(const bf16* __restrict__ qg,
                            const bf16* __restrict__ kg,
                            const bf16* __restrict__ vg,
                            bf16* __restrict__ ao) {
  int blk = blockIdx.x;
  int rt = blk & 15, h = (blk >> 4) & 7, b = blk >> 7;
  size_t bh = ((size_t)b * HEADS + h) * NSP * HD;
  const bf16* Q = qg + bh;
  const bf16* K = kg + bh;
  const bf16* V = vg + bh;
  int n0 = rt * 64;
  int t = threadIdx.x, lane = t & 31, wave = t >> 5;
  int l15 = lane & 15, half = lane >> 4;
  __shared__ __align__(16) bf16 QS[64][72];
  __shared__ __align__(16) bf16 KS[32][72];
  __shared__ __align__(16) bf16 VTS[64][40];
  __shared__ __align__(16) bf16 PS[4][16][40];
#if USE_ASYNC
  {  // async stage Q block (64 x 64): 512 x 16B chunks
    const bf16* gq = Q + (size_t)n0 * HD;
#pragma unroll
    for (int i = 0; i < 4; ++i) {
      int c = i * 128 + t;
      int r = c >> 3, qd = c & 7;
      async_cp16(gq + (size_t)r * HD + qd * 8, &QS[r][qd * 8]);
    }
    __builtin_amdgcn_s_wait_asynccnt(0);
  }
#else
  {  // stage Q block (64 x 64)
    const uint32_t* src = (const uint32_t*)(Q + (size_t)n0 * HD);
#pragma unroll
    for (int i = 0; i < 16; ++i) {
      int flat = i * 128 + t;
      int r = flat >> 5, cd = flat & 31;
      ((uint32_t*)&QS[r][0])[cd] = src[(size_t)r * 32 + cd];
    }
  }
#endif
  __syncthreads();
  union bfu { bf16x16 v; uint4 u[2]; };
  bfu qa[2];
  {
    int ah = half * 16;
    int am = wave * 16 + l15;
#pragma unroll
    for (int kc = 0; kc < 2; ++kc) {
      qa[kc].u[0] = *(const uint4*)&QS[am][kc * 32 + ah];
      qa[kc].u[1] = *(const uint4*)&QS[am][kc * 32 + ah + 8];
    }
  }
  float mi[8], li[8];
#pragma unroll
  for (int r = 0; r < 8; ++r) { mi[r] = -1e30f; li[r] = 0.f; }
  f32x8 acc[4] = {};
  const float scale = 0.125f;  // hd^-0.5 (power of 2; folded into exp args)

  for (int mt0 = 0; mt0 < NSP; mt0 += 32) {
    __syncthreads();
#if USE_ASYNC
    {  // async stage K tile (32 x 64): 256 x 16B chunks
      const bf16* gk = K + (size_t)mt0 * HD;
#pragma unroll
      for (int i = 0; i < 2; ++i) {
        int c = i * 128 + t;
        int r = c >> 3, qd = c & 7;
        async_cp16(gk + (size_t)r * HD + qd * 8, &KS[r][qd * 8]);
      }
    }
#else
    {  // stage K tile (32 x 64)
      const uint32_t* src = (const uint32_t*)(K + (size_t)mt0 * HD);
#pragma unroll
      for (int i = 0; i < 8; ++i) {
        int flat = i * 128 + t;
        int r = flat >> 5, cd = flat & 31;
        ((uint32_t*)&KS[r][0])[cd] = src[(size_t)r * 32 + cd];
      }
    }
#endif
    {  // stage V transposed: VTS[d][m]
      const uint32_t* src = (const uint32_t*)(V + (size_t)mt0 * HD);
#pragma unroll
      for (int i = 0; i < 8; ++i) {
        int flat = i * 128 + t;
        int m = flat >> 5, cd = flat & 31;
        union { uint32_t u; bf16 hh[2]; } pr;
        pr.u = src[(size_t)m * 32 + cd];
        VTS[cd * 2 + 0][m] = pr.hh[0];
        VTS[cd * 2 + 1][m] = pr.hh[1];
      }
    }
#if USE_ASYNC
    __builtin_amdgcn_s_wait_asynccnt(0);
#endif
    __syncthreads();
    // S = Q K^T  (16 rows x 32 cols per wave)
    f32x8 s0 = {}, s1 = {};
    int bh2 = half * 16;
#pragma unroll
    for (int kc = 0; kc < 2; ++kc) {
      bfu B0, B1;
      const uint4* p0 = (const uint4*)&KS[l15][kc * 32 + bh2];
      const uint4* p1 = (const uint4*)&KS[16 + l15][kc * 32 + bh2];
      B0.u[0] = p0[0]; B0.u[1] = p0[1];
      B1.u[0] = p1[0]; B1.u[1] = p1[1];
      s0 = __builtin_amdgcn_wmma_f32_16x16x32_bf16(false, qa[kc].v, false, B0.v, (short)0, s0, false, false);
      s1 = __builtin_amdgcn_wmma_f32_16x16x32_bf16(false, qa[kc].v, false, B1.v, (short)0, s1, false, false);
    }
    // online softmax in unscaled units (lane holds col l15, rows r+8*half)
    float rmax[8];
#pragma unroll
    for (int r = 0; r < 8; ++r) rmax[r] = fmaxf(s0[r], s1[r]);
#pragma unroll
    for (int r = 0; r < 8; ++r)
#pragma unroll
      for (int m = 1; m <= 8; m <<= 1)
        rmax[r] = fmaxf(rmax[r], __shfl_xor(rmax[r], m, 32));
    float p0v[8], p1v[8], rsum[8];
#pragma unroll
    for (int r = 0; r < 8; ++r) {
      float nm = fmaxf(mi[r], rmax[r]);
      float al = __expf((mi[r] - nm) * scale);
      p0v[r] = __expf((s0[r] - nm) * scale);
      p1v[r] = __expf((s1[r] - nm) * scale);
      rsum[r] = p0v[r] + p1v[r];
      mi[r] = nm;
      li[r] *= al;
#pragma unroll
      for (int g2 = 0; g2 < 4; ++g2) acc[g2][r] *= al;
    }
#pragma unroll
    for (int r = 0; r < 8; ++r)
#pragma unroll
      for (int m = 1; m <= 8; m <<= 1)
        rsum[r] += __shfl_xor(rsum[r], m, 32);
#pragma unroll
    for (int r = 0; r < 8; ++r) li[r] += rsum[r];
    // P -> per-wave LDS slab (C-layout -> A-layout round trip)
#pragma unroll
    for (int r = 0; r < 8; ++r) {
      int row = r + half * 8;
      PS[wave][row][l15] = tobf(p0v[r]);
      PS[wave][row][16 + l15] = tobf(p1v[r]);
    }
    // O += P V
    {
      bfu Pa;
      int ah = half * 16;
      Pa.u[0] = *(const uint4*)&PS[wave][l15][ah];
      Pa.u[1] = *(const uint4*)&PS[wave][l15][ah + 8];
#pragma unroll
      for (int g2 = 0; g2 < 4; ++g2) {
        bfu Bv;
        const uint4* p = (const uint4*)&VTS[g2 * 16 + l15][bh2];
        Bv.u[0] = p[0]; Bv.u[1] = p[1];
        acc[g2] = __builtin_amdgcn_wmma_f32_16x16x32_bf16(
            false, Pa.v, false, Bv.v, (short)0, acc[g2], false, false);
      }
    }
  }
  // epilogue: ao[b][n][h*64+d] bf16
#pragma unroll
  for (int g2 = 0; g2 < 4; ++g2) {
    int d = g2 * 16 + l15;
#pragma unroll
    for (int r = 0; r < 8; ++r) {
      int n = n0 + wave * 16 + r + half * 8;
      float ov = acc[g2][r] / li[r];
      ao[((size_t)b * NSP + n) * CCH + h * HD + d] = tobf(ov);
    }
  }
}

// ---------------------------------------------------------------- out proj
// grid 1024 = b(8) * mt(16) * ot(8); block 128
__global__ void proj_gemm(const bf16* __restrict__ ain, const float* __restrict__ wp,
                          const float* __restrict__ bias, const float* __restrict__ x,
                          float* __restrict__ out) {
  int blk = blockIdx.x;
  int ot = blk & 7, mt = (blk >> 3) & 15, b = blk >> 7;
  int o0 = ot * 64, m0 = mt * 64;
  __shared__ __align__(16) bf16 XS[64][40];
  __shared__ __align__(16) bf16 WS[64][40];
  int t = threadIdx.x, lane = t & 31, wave = t >> 5;
  int l15 = lane & 15, half = lane >> 4;
  f32x8 acc[4] = {};
  const bf16* xrow = ain + ((size_t)b * NSP + m0) * CCH;
  for (int kt = 0; kt < 16; ++kt) {
    __syncthreads();
#if USE_ASYNC
    {
#pragma unroll
      for (int i = 0; i < 2; ++i) {
        int c = i * 128 + t;
        int r = c >> 2, qd = c & 3;
        async_cp16(xrow + (size_t)r * CCH + kt * 32 + qd * 8, &XS[r][qd * 8]);
      }
    }
#else
    {
      const uint32_t* src = (const uint32_t*)xrow + kt * 16;
#pragma unroll
      for (int i = 0; i < 8; ++i) {
        int flat = i * 128 + t;
        int r = flat >> 4, cd = flat & 15;
        ((uint32_t*)&XS[r][0])[cd] = src[(size_t)r * 256 + cd];
      }
    }
#endif
    {
      const float* wsrc = wp + (size_t)o0 * CCH + kt * 32;
#pragma unroll
      for (int i = 0; i < 8; ++i) {
        int flat = i * 128 + t;
        int r = flat >> 4, cp = flat & 15;
        float2 pv = *(const float2*)(wsrc + (size_t)r * CCH + cp * 2);
        ((uint32_t*)&WS[r][0])[cp] = packbf(pv.x, pv.y);
      }
    }
#if USE_ASYNC
    __builtin_amdgcn_s_wait_asynccnt(0);
#endif
    __syncthreads();
    int ah = half * 16;
    int am = wave * 16 + l15;
    union { bf16x16 v; uint4 u[2]; } A;
    A.u[0] = *(const uint4*)&XS[am][ah];
    A.u[1] = *(const uint4*)&XS[am][ah + 8];
    int bh = half * 16;
#pragma unroll
    for (int grp = 0; grp < 4; ++grp) {
      union { bf16x16 v; uint4 u[2]; } Bm;
      const uint4* p = (const uint4*)&WS[grp * 16 + l15][bh];
      Bm.u[0] = p[0]; Bm.u[1] = p[1];
      acc[grp] = __builtin_amdgcn_wmma_f32_16x16x32_bf16(
          false, A.v, false, Bm.v, (short)0, acc[grp], false, false);
    }
  }
  // epilogue: out[b][o][n] = x + proj + bias  (fp32)
#pragma unroll
  for (int grp = 0; grp < 4; ++grp) {
    int o = o0 + grp * 16 + l15;
    float bo = bias[o];
#pragma unroll
    for (int r = 0; r < 8; ++r) {
      int n = m0 + wave * 16 + r + half * 8;
      size_t off = ((size_t)b * CCH + o) * NSP + n;
      out[off] = x[off] + acc[grp][r] + bo;
    }
  }
}

// ---------------------------------------------------------------- launch
extern "C" void kernel_launch(void* const* d_in, const int* in_sizes, int n_in,
                              void* d_out, int out_size, void* d_ws, size_t ws_size,
                              hipStream_t stream) {
  const float* x      = (const float*)d_in[0];
  const float* gamma  = (const float*)d_in[1];
  const float* beta   = (const float*)d_in[2];
  const float* w_qkv  = (const float*)d_in[3];
  const float* w_proj = (const float*)d_in[4];
  const float* b_proj = (const float*)d_in[5];
  float* out = (float*)d_out;

  char* ws = (char*)d_ws;
  const size_t SZ = (size_t)BB * NSP * CCH * 2;  // 8 MiB per bf16 tensor
  float* stats = (float*)ws;                     // 512 B (+pad)
  bf16* xn = (bf16*)(ws + 1024);                 // [b][n][c]
  bf16* qb = (bf16*)(ws + 1024 + SZ);            // [b][h][n][d]
  bf16* kb = (bf16*)(ws + 1024 + 2 * SZ);
  bf16* vb = (bf16*)(ws + 1024 + 3 * SZ);
  bf16* ao = xn;  // xn dead after qkv_gemm; reuse for attn output

  zero_stats<<<1, 128, 0, stream>>>(stats);
  gn_reduce<<<512, 256, 0, stream>>>(x, stats);
  gn_norm<<<1024, 256, 0, stream>>>(x, gamma, beta, stats, xn);
  qkv_gemm<<<8 * 16 * 24, 128, 0, stream>>>(xn, w_qkv, qb, kb, vb);
  attn_kernel<<<8 * 8 * 16, 128, 0, stream>>>(qb, kb, vb, ao);
  proj_gemm<<<8 * 16 * 8, 128, 0, stream>>>(ao, w_proj, b_proj, x, out);
}
